// RoPEViT_49091476193515
// MI455X (gfx1250) — matrix operand order repaired and checked by reading, not compiled, not run on hardware
//
#include <hip/hip_runtime.h>
#include <hip/hip_bf16.h>
#include <cmath>

typedef __bf16 bf16;
typedef __attribute__((ext_vector_type(16))) __bf16 v16bf;
typedef __attribute__((ext_vector_type(8)))  __bf16 v8bf;
typedef __attribute__((ext_vector_type(8)))  float  v8f;

#define DEPTH   12
#define HEADS   12
#define DIM     768
#define DH      64
#define MLP_DIM 3072
#define BATCH   32
#define SEQ     197
#define NPATCH  196
#define NPAD    224
#define MROWS   (BATCH*SEQ)     // 6304
#define PROWS   (BATCH*NPATCH)  // 6272
#define BH      (BATCH*HEADS)   // 384

#define BM 64
#define BN 128
#define BK 32
#define BKP 40   // padded LDS K-stride (bf16); 80B row stride, 16B aligned

enum { EPI_BIAS = 1, EPI_GELU = 2, EPI_RESID = 4 };

static const int BIGDIV = 0x40000000;

// ---------------------------------------------------------------------------
// CDNA5 async copy: 16B global -> LDS, tracked by ASYNCcnt (no VGPR staging).
// ---------------------------------------------------------------------------
__device__ __forceinline__ void async_ld16(const bf16* gsrc, bf16* ldst)
{
    unsigned lo = (unsigned)(uintptr_t)ldst;   // low 32 bits of flat addr = LDS offset
    asm volatile("global_load_async_to_lds_b128 %0, %1, off"
                 :: "v"(lo), "v"(gsrc) : "memory");
}
// steady state: 3 newer async ops (next tile) may remain outstanding
__device__ __forceinline__ void wait_async3()
{
    asm volatile("s_wait_asynccnt 0x3" ::: "memory");
}
__device__ __forceinline__ void wait_async0()
{
    asm volatile("s_wait_asynccnt 0x0" ::: "memory");
}

// ---------------------------------------------------------------------------
// One BK-step of WMMA over an LDS tile pair (ISA 7.12.2 lane layouts).
// ---------------------------------------------------------------------------
__device__ __forceinline__ void wmma_tile(const bf16 (*lA)[BKP], const bf16 (*lB)[BKP],
                                          int wy, int wx, int r, int half,
                                          v8f (&acc)[2][2])
{
    union Frag { v16bf v; v8bf h[2]; };
    Frag af[2], bfg[2];
    for (int i = 0; i < 2; ++i) {
        const int rowA = wy*32 + i*16 + r;
        af[i].h[0] = *(const v8bf*)&lA[rowA][half*8];
        af[i].h[1] = *(const v8bf*)&lA[rowA][16 + half*8];
        const int colB = wx*32 + i*16 + r;
        bfg[i].h[0] = *(const v8bf*)&lB[colB][half*8];
        bfg[i].h[1] = *(const v8bf*)&lB[colB][16 + half*8];
    }
    for (int i = 0; i < 2; ++i)
        for (int j = 0; j < 2; ++j)
            acc[i][j] = __builtin_amdgcn_wmma_f32_16x16x32_bf16(
                false, af[i].v, false, bfg[j].v,
                (short)0, acc[i][j], false, false);
}

// ---------------------------------------------------------------------------
// bf16 WMMA GEMM:  C = epi( scale * A[M,K] x B^T + bias [, + resid] )
//  - A bf16 row-major [M,K], batch offset z*sA
//  - B bf16 [N,K] row-major (pre-transposed), batch offset z*sB
//  - C fp32 (OUTBF16=0, resid allowed) or bf16 (OUTBF16=1)
//  - batch C offset (z/divC)*sC1 + (z%divC)*sC2 ; K % 32 == 0 required
// Interior blocks: double-buffered async-to-LDS pipeline; edge blocks: guarded.
// ---------------------------------------------------------------------------
template<int EPI, bool OUTBF16>
__global__ __launch_bounds__(256)
void gemm_bf16t(const bf16* __restrict__ A, long sA,
                const bf16* __restrict__ Bt, long sB,
                void* __restrict__ Cv, int divC, long sC1, long sC2,
                const float* __restrict__ bias,
                const float* __restrict__ resid,
                int M, int N, int K, int ldc, float scale)
{
    __shared__ alignas(16) bf16 ldsA[2][BM][BKP];
    __shared__ alignas(16) bf16 ldsB[2][BN][BKP];

    const int t    = threadIdx.x;
    const int lane = t & 31;
    const int wave = t >> 5;
    const int wy   = wave >> 2;       // 0..1
    const int wx   = wave & 3;        // 0..3
    const int r    = lane & 15;
    const int half = lane >> 4;

    const int z    = blockIdx.z;
    const int row0 = blockIdx.y * BM;
    const int col0 = blockIdx.x * BN;

    const bf16* Ab = A  + (long)z * sA;
    const bf16* Bb = Bt + (long)z * sB;
    const long offC = (long)(z / divC) * sC1 + (long)(z % divC) * sC2;

    // staging coords: A tile 64x32 bf16 = 256 x 16B; B tile 128x32 = 512 x 16B
    const int ar  = t >> 2,      ak  = (t & 3) * 8;
    const int bn0 = (t*2) >> 2,  bk0 = ((t*2) & 3) * 8;
    const int bn1 = (t*2+1) >> 2,bk1 = ((t*2+1) & 3) * 8;
    const int gra = row0 + ar;

    const v8f vz = {0.f,0.f,0.f,0.f,0.f,0.f,0.f,0.f};
    v8f acc[2][2];
    acc[0][0]=vz; acc[0][1]=vz; acc[1][0]=vz; acc[1][1]=vz;

    const bool interior = (row0 + BM <= M) && (col0 + BN <= N);

    if (interior) {
        // -------- pipelined async path (branch-free hot loop) --------
        const bf16* gA  = Ab + (long)gra * K + ak;
        const bf16* gB0 = Bb + (long)(col0 + bn0) * K + bk0;
        const bf16* gB1 = Bb + (long)(col0 + bn1) * K + bk1;

        // prologue: tile 0 -> buffer 0
        async_ld16(gA,  &ldsA[0][ar][ak]);
        async_ld16(gB0, &ldsB[0][bn0][bk0]);
        async_ld16(gB1, &ldsB[0][bn1][bk1]);

        int buf = 0;
        for (int k0 = 0; k0 < K - BK; k0 += BK) {
            const int kn = k0 + BK;
            async_ld16(gA  + kn, &ldsA[buf^1][ar][ak]);
            async_ld16(gB0 + kn, &ldsB[buf^1][bn0][bk0]);
            async_ld16(gB1 + kn, &ldsB[buf^1][bn1][bk1]);
            wait_async3();              // oldest 3 (tile k0) complete
            __syncthreads();
            wmma_tile(ldsA[buf], ldsB[buf], wy, wx, r, half, acc);
            __syncthreads();            // all waves done reading before buf is re-targeted
            buf ^= 1;
        }
        // tail tile
        wait_async0();
        __syncthreads();
        wmma_tile(ldsA[buf], ldsB[buf], wy, wx, r, half, acc);
    } else {
        // -------- guarded fallback (edge blocks only) --------
        const bool aok = (gra < M);
        const bool b0ok = (col0 + bn0 < N);
        const bool b1ok = (col0 + bn1 < N);
        for (int k0 = 0; k0 < K; k0 += BK) {
            uint4 v = {0u,0u,0u,0u};
            if (aok) v = *(const uint4*)(Ab + (long)gra * K + k0 + ak);
            *(uint4*)&ldsA[0][ar][ak] = v;
            uint4 w0 = {0u,0u,0u,0u};
            if (b0ok) w0 = *(const uint4*)(Bb + (long)(col0 + bn0) * K + k0 + bk0);
            *(uint4*)&ldsB[0][bn0][bk0] = w0;
            uint4 w1 = {0u,0u,0u,0u};
            if (b1ok) w1 = *(const uint4*)(Bb + (long)(col0 + bn1) * K + k0 + bk1);
            *(uint4*)&ldsB[0][bn1][bk1] = w1;
            __syncthreads();
            wmma_tile(ldsA[0], ldsB[0], wy, wx, r, half, acc);
            __syncthreads();
        }
    }

    // ---- epilogue (C/D layout: vgpr e -> M = e + half*8, lane r -> N) ----
    for (int i = 0; i < 2; ++i) {
        for (int j = 0; j < 2; ++j) {
            const int gc = col0 + wx*32 + j*16 + r;
            if (gc >= N) continue;
            for (int e = 0; e < 8; ++e) {
                const int gr = row0 + wy*32 + i*16 + half*8 + e;
                if (gr >= M) continue;
                float v = acc[i][j][e] * scale;
                if (EPI & EPI_BIAS)  v += bias[gc];
                if (EPI & EPI_GELU)  v = 0.5f * v * (1.0f + erff(v * 0.70710678118654752f));
                const long co = offC + (long)gr * ldc + gc;
                if (OUTBF16) {
                    ((bf16*)Cv)[co] = (bf16)v;
                } else {
                    if (EPI & EPI_RESID) v += resid[co];
                    ((float*)Cv)[co] = v;
                }
            }
        }
    }
}

// ---------------------------------------------------------------------------
// Transpose + convert: W[K,N] fp32 row-major -> Wt[N,K] bf16 row-major.
// ---------------------------------------------------------------------------
__global__ __launch_bounds__(256)
void transpose_to_bf16(const float* __restrict__ W, long sW,
                       bf16* __restrict__ Wt, long sWt, int K, int N)
{
    __shared__ float tile[32][33];
    const int z = blockIdx.z;
    const float* Wz = W + (long)z * sW;
    bf16* Wtz = Wt + (long)z * sWt;
    const int n0 = blockIdx.x * 32, k0 = blockIdx.y * 32;
    const int tx = threadIdx.x & 31, ty = threadIdx.x >> 5;
    for (int i = 0; i < 32; i += 8) {
        const int k = k0 + ty + i, n = n0 + tx;
        tile[ty + i][tx] = (k < K && n < N) ? Wz[(long)k * N + n] : 0.0f;
    }
    __syncthreads();
    for (int i = 0; i < 32; i += 8) {
        const int n = n0 + ty + i, k = k0 + tx;
        if (n < N && k < K) Wtz[(long)n * K + k] = (bf16)tile[tx][ty + i];
    }
}

// ---------------------------------------------------------------------------
// Row LayerNorm (D <= 768); fp32 in, fp32 or bf16 out.
// offset(row) = (row/div)*s1 + (row%div)*s2
// ---------------------------------------------------------------------------
template<bool OUTBF16>
__global__ __launch_bounds__(256)
void ln_rows(const float* __restrict__ in, int divI, long sI1, long sI2,
             void* __restrict__ outv, int divO, long sO1, long sO2,
             const float* __restrict__ w, const float* __restrict__ b, int D)
{
    __shared__ float vals[768];
    __shared__ float sred[2][8];
    const int row = blockIdx.x;
    const int t = threadIdx.x, lane = t & 31, wave = t >> 5;
    const float* src = in + (long)(row / divI) * sI1 + (long)(row % divI) * sI2;

    float s = 0.f, ss = 0.f;
    for (int f = t; f < D; f += 256) {
        float v = src[f]; vals[f] = v; s += v; ss += v * v;
    }
    for (int m = 16; m > 0; m >>= 1) { s += __shfl_xor(s, m, 32); ss += __shfl_xor(ss, m, 32); }
    if (lane == 0) { sred[0][wave] = s; sred[1][wave] = ss; }
    __syncthreads();
    if (t == 0) {
        float a = 0.f, c = 0.f;
        for (int i = 0; i < 8; ++i) { a += sred[0][i]; c += sred[1][i]; }
        sred[0][0] = a; sred[1][0] = c;
    }
    __syncthreads();
    const float mean = sred[0][0] / (float)D;
    const float var  = sred[1][0] / (float)D - mean * mean;
    const float rstd = rsqrtf(var + 1e-5f);
    const long ob = (long)(row / divO) * sO1 + (long)(row % divO) * sO2;
    for (int f = t; f < D; f += 256) {
        float v = (vals[f] - mean) * rstd * w[f] + b[f];
        if (OUTBF16) ((bf16*)outv)[ob + f] = (bf16)v;
        else         ((float*)outv)[ob + f] = v;
    }
}

// ---------------------------------------------------------------------------
// Patch extraction + LN1 -> bf16 patch vectors [B*196, 768].
// feature f = (iy*16+ix)*3 + ch  <-  img[b, ch, py*16+iy, px*16+ix]
// ---------------------------------------------------------------------------
__global__ __launch_bounds__(256)
void patch_extract_ln(const float* __restrict__ img,
                      const float* __restrict__ w, const float* __restrict__ b,
                      bf16* __restrict__ pin)
{
    __shared__ float vals[768];
    __shared__ float sred[2][8];
    const int row = blockIdx.x;
    const int bi = row / NPATCH, n = row % NPATCH;
    const int py = n / 14, px = n % 14;
    const int t = threadIdx.x, lane = t & 31, wave = t >> 5;

    float s = 0.f, ss = 0.f;
    for (int f = t; f < 768; f += 256) {
        const int pi = f / 3, ch = f % 3;
        const int iy = pi >> 4, ix = pi & 15;
        float v = img[(((long)bi * 3 + ch) * 224 + py * 16 + iy) * 224 + px * 16 + ix];
        vals[f] = v; s += v; ss += v * v;
    }
    for (int m = 16; m > 0; m >>= 1) { s += __shfl_xor(s, m, 32); ss += __shfl_xor(ss, m, 32); }
    if (lane == 0) { sred[0][wave] = s; sred[1][wave] = ss; }
    __syncthreads();
    if (t == 0) {
        float a = 0.f, c = 0.f;
        for (int i = 0; i < 8; ++i) { a += sred[0][i]; c += sred[1][i]; }
        sred[0][0] = a; sred[1][0] = c;
    }
    __syncthreads();
    const float mean = sred[0][0] / 768.0f;
    const float var  = sred[1][0] / 768.0f - mean * mean;
    const float rstd = rsqrtf(var + 1e-5f);
    for (int f = t; f < 768; f += 256)
        pin[(long)row * 768 + f] = (bf16)((vals[f] - mean) * rstd * w[f] + b[f]);
}

// ---------------------------------------------------------------------------
// RoPE + repack bf16 qkv[B*197, 2304] ->
//   Qp/Kp [B*H, NPAD, 64]  (token-major, zero padded)
//   Vt    [B*H, 64, NPAD]  (transposed for B^T GEMM, zero padded)
// ---------------------------------------------------------------------------
__global__ __launch_bounds__(256)
void rope_pack(const bf16* __restrict__ qkv,
               bf16* __restrict__ Qp, bf16* __restrict__ Kp, bf16* __restrict__ Vt)
{
    long idx = (long)blockIdx.x * 256 + threadIdx.x;
    const long total = (long)BH * NPAD * (DH / 2);
    if (idx >= total) return;
    const int p = (int)(idx & 31);
    long rr = idx >> 5;
    const int i = (int)(rr % NPAD);
    const int zh = (int)(rr / NPAD);          // b*HEADS + h
    const long qko = ((long)zh * NPAD + i) * DH + 2 * p;
    const long vto0 = ((long)zh * DH + 2 * p) * NPAD + i;
    const long vto1 = vto0 + NPAD;

    if (i >= SEQ) {
        Qp[qko] = (bf16)0.f; Qp[qko + 1] = (bf16)0.f;
        Kp[qko] = (bf16)0.f; Kp[qko + 1] = (bf16)0.f;
        Vt[vto0] = (bf16)0.f; Vt[vto1] = (bf16)0.f;
        return;
    }
    const int b = zh / HEADS, h = zh % HEADS;
    const bf16* src = qkv + ((long)(b * SEQ + i)) * (3 * DIM) + h * DH + 2 * p;
    float qe = (float)src[0],         qo = (float)src[1];
    float ke = (float)src[DIM],       ko = (float)src[DIM + 1];
    float ve = (float)src[2 * DIM],   vo = (float)src[2 * DIM + 1];

    if (i >= 1) {  // cls (i==0): no RoPE
        const int tok = i - 1;
        const int j   = (p < 16) ? p : (p - 16);
        const float pos = (p < 16) ? (float)(tok % 14) : (float)(tok / 14);
        const float freq = __powf(100.0f, -(float)j * (1.0f / 16.0f));
        float sn, cs;
        __sincosf(pos * freq, &sn, &cs);
        float qre = qe * cs - qo * sn, qim = qe * sn + qo * cs;
        float kre = ke * cs - ko * sn, kim = ke * sn + ko * cs;
        qe = qre; qo = qim; ke = kre; ko = kim;
    }
    Qp[qko] = (bf16)qe; Qp[qko + 1] = (bf16)qo;
    Kp[qko] = (bf16)ke; Kp[qko + 1] = (bf16)ko;
    Vt[vto0] = (bf16)ve; Vt[vto1] = (bf16)vo;
}

// ---------------------------------------------------------------------------
// Softmax over rows of S[B*H, NPAD, NPAD] fp32 -> P bf16 (masked to 197).
// One wave per row (7 elems / lane).
// ---------------------------------------------------------------------------
__global__ __launch_bounds__(256)
void softmax_rows(const float* __restrict__ S, bf16* __restrict__ P)
{
    const int z = blockIdx.y;
    const int wave = threadIdx.x >> 5, lane = threadIdx.x & 31;
    const int rrow = blockIdx.x * 8 + wave;
    if (rrow >= SEQ) return;
    const long ro = ((long)z * NPAD + rrow) * NPAD;

    float v[7];
    float vmax = -3.0e38f;
    for (int q = 0; q < 7; ++q) {
        const int j = lane + q * 32;
        float x = (j < SEQ) ? S[ro + j] : -3.0e38f;
        v[q] = x; vmax = fmaxf(vmax, x);
    }
    for (int m = 16; m > 0; m >>= 1) vmax = fmaxf(vmax, __shfl_xor(vmax, m, 32));
    float sum = 0.f;
    for (int q = 0; q < 7; ++q) {
        const int j = lane + q * 32;
        float e = (j < SEQ) ? __expf(v[q] - vmax) : 0.0f;
        v[q] = e; sum += e;
    }
    for (int m = 16; m > 0; m >>= 1) sum += __shfl_xor(sum, m, 32);
    const float inv = 1.0f / sum;
    for (int q = 0; q < 7; ++q) {
        const int j = lane + q * 32;
        P[ro + j] = (bf16)(v[q] * inv);
    }
}

// ---------------------------------------------------------------------------
__global__ void set_cls(const float* __restrict__ cls, float* __restrict__ x)
{
    const int b = blockIdx.x;
    for (int f = threadIdx.x; f < DIM; f += 256)
        x[(long)b * SEQ * DIM + f] = cls[f];
}

// ---------------------------------------------------------------------------
extern "C" void kernel_launch(void* const* d_in, const int* in_sizes, int n_in,
                              void* d_out, int out_size, void* d_ws, size_t ws_size,
                              hipStream_t stream)
{
    const float* img       = (const float*)d_in[0];
    const float* p_ln1_w   = (const float*)d_in[1];
    const float* p_ln1_b   = (const float*)d_in[2];
    const float* patch_w   = (const float*)d_in[3];
    const float* patch_b   = (const float*)d_in[4];
    const float* p_ln2_w   = (const float*)d_in[5];
    const float* p_ln2_b   = (const float*)d_in[6];
    const float* cls_token = (const float*)d_in[7];
    const float* attn_ln_w = (const float*)d_in[8];
    const float* attn_ln_b = (const float*)d_in[9];
    const float* w_qkv     = (const float*)d_in[10];
    const float* w_out     = (const float*)d_in[11];
    const float* b_out     = (const float*)d_in[12];
    const float* ff_ln_w   = (const float*)d_in[13];
    const float* ff_ln_b   = (const float*)d_in[14];
    const float* w1        = (const float*)d_in[15];
    const float* b1        = (const float*)d_in[16];
    const float* w2        = (const float*)d_in[17];
    const float* b2        = (const float*)d_in[18];
    const float* head_ln_w = (const float*)d_in[19];
    const float* head_ln_b = (const float*)d_in[20];
    const float* head_w    = (const float*)d_in[21];
    const float* head_b    = (const float*)d_in[22];
    float* out = (float*)d_out;

    // ---- workspace layout (256B aligned bump allocator) ----
    char* base = (char*)d_ws;
    auto alloc = [&](long bytes) { char* p = base; base += (bytes + 255) & ~255L; return p; };
    const long SZ_X = (long)MROWS * DIM;

    float* x    = (float*)alloc(SZ_X * 4);                       // residual stream
    float* S    = (float*)alloc((long)BH * NPAD * NPAD * 4);     // scores (also patch-GEMM fp32 scratch)
    bf16*  h    = (bf16*) alloc(SZ_X * 2);                       // LN out (also pin / hcls)
    bf16*  qkvb = (bf16*) alloc((long)MROWS * 3 * DIM * 2);
    bf16*  Qp   = (bf16*) alloc((long)BH * NPAD * DH * 2);
    bf16*  Kp   = (bf16*) alloc((long)BH * NPAD * DH * 2);
    bf16*  Vt   = (bf16*) alloc((long)BH * DH * NPAD * 2);
    bf16*  o    = (bf16*) alloc(SZ_X * 2);                       // attention output
    bf16*  PH   = (bf16*) alloc((long)MROWS * MLP_DIM * 2);      // P (bf16 probs) aliases MLP hidden
    bf16*  P    = PH;
    bf16*  hid  = PH;
    // transposed bf16 weights
    bf16* patch_wT = (bf16*)alloc((long)DIM * DIM * 2);
    bf16* w_qkvT   = (bf16*)alloc((long)DEPTH * DIM * 3 * DIM * 2);
    bf16* w_outT   = (bf16*)alloc((long)DEPTH * DIM * DIM * 2);
    bf16* w1T      = (bf16*)alloc((long)DEPTH * DIM * MLP_DIM * 2);
    bf16* w2T      = (bf16*)alloc((long)DEPTH * MLP_DIM * DIM * 2);
    bf16* head_wT  = (bf16*)alloc((long)DIM * 1000 * 2);

    auto cdiv = [](int a, int b) { return (a + b - 1) / b; };

    // ---- one-time weight transpose+convert to bf16 [N,K] ----
    transpose_to_bf16<<<dim3(cdiv(DIM,32),  cdiv(DIM,32),  1),     256, 0, stream>>>(patch_w, 0, patch_wT, 0, DIM, DIM);
    transpose_to_bf16<<<dim3(cdiv(3*DIM,32),cdiv(DIM,32),  DEPTH), 256, 0, stream>>>(w_qkv, (long)DIM*3*DIM, w_qkvT, (long)DIM*3*DIM, DIM, 3*DIM);
    transpose_to_bf16<<<dim3(cdiv(DIM,32),  cdiv(DIM,32),  DEPTH), 256, 0, stream>>>(w_out, (long)DIM*DIM, w_outT, (long)DIM*DIM, DIM, DIM);
    transpose_to_bf16<<<dim3(cdiv(MLP_DIM,32),cdiv(DIM,32),DEPTH), 256, 0, stream>>>(w1, (long)DIM*MLP_DIM, w1T, (long)DIM*MLP_DIM, DIM, MLP_DIM);
    transpose_to_bf16<<<dim3(cdiv(DIM,32),  cdiv(MLP_DIM,32),DEPTH),256,0, stream>>>(w2, (long)MLP_DIM*DIM, w2T, (long)MLP_DIM*DIM, MLP_DIM, DIM);
    transpose_to_bf16<<<dim3(cdiv(1000,32), cdiv(DIM,32),  1),     256, 0, stream>>>(head_w, 0, head_wT, 0, DIM, 1000);

    // ---- patch embedding ----
    patch_extract_ln<<<PROWS, 256, 0, stream>>>(img, p_ln1_w, p_ln1_b, h /*pin*/);
    {
        dim3 g(cdiv(DIM, BN), cdiv(PROWS, BM), 1);
        gemm_bf16t<EPI_BIAS, false><<<g, 256, 0, stream>>>(
            h, 0, patch_wT, 0, S /*fp32 scratch*/, 1, 0, 0, patch_b, nullptr,
            PROWS, DIM, DIM, DIM, 1.0f);
    }
    ln_rows<false><<<PROWS, 256, 0, stream>>>(S, BIGDIV, 0, DIM,
                                              x + DIM, NPATCH, (long)SEQ * DIM, DIM,
                                              p_ln2_w, p_ln2_b, DIM);
    set_cls<<<BATCH, 256, 0, stream>>>(cls_token, x);

    // ---- transformer layers ----
    for (int L = 0; L < DEPTH; ++L) {
        ln_rows<true><<<MROWS, 256, 0, stream>>>(x, BIGDIV, 0, DIM,
                                                 h, BIGDIV, 0, DIM,
                                                 attn_ln_w + (long)L * DIM, attn_ln_b + (long)L * DIM, DIM);
        {   // QKV
            dim3 g(cdiv(3 * DIM, BN), cdiv(MROWS, BM), 1);
            gemm_bf16t<0, true><<<g, 256, 0, stream>>>(
                h, 0, w_qkvT + (long)L * DIM * 3 * DIM, 0, qkvb, 1, 0, 0,
                nullptr, nullptr, MROWS, 3 * DIM, DIM, 3 * DIM, 1.0f);
        }
        {   // RoPE + repack
            const long total = (long)BH * NPAD * (DH / 2);
            rope_pack<<<(int)((total + 255) / 256), 256, 0, stream>>>(qkvb, Qp, Kp, Vt);
        }
        {   // S = scale * Q x K^T   (batched over B*H)
            dim3 g(cdiv(NPAD, BN), cdiv(NPAD, BM), BH);
            gemm_bf16t<0, false><<<g, 256, 0, stream>>>(
                Qp, (long)NPAD * DH, Kp, (long)NPAD * DH,
                S, 1, (long)NPAD * NPAD, 0,
                nullptr, nullptr, NPAD, NPAD, DH, NPAD, 0.125f);
        }
        {   // softmax -> bf16 P
            dim3 g(cdiv(SEQ, 8), BH);
            softmax_rows<<<g, 256, 0, stream>>>(S, P);
        }
        {   // O = P x V   (B = V^T [64, 224]); strided bf16 out o[b, i, h*64+d]
            dim3 g(cdiv(DH, BN), cdiv(SEQ, BM), BH);
            gemm_bf16t<0, true><<<g, 256, 0, stream>>>(
                P, (long)NPAD * NPAD, Vt, (long)DH * NPAD,
                o, HEADS, (long)SEQ * DIM, (long)DH,
                nullptr, nullptr, SEQ, DH, NPAD, DIM, 1.0f);
        }
        {   // out projection + bias + residual -> x
            dim3 g(cdiv(DIM, BN), cdiv(MROWS, BM), 1);
            gemm_bf16t<EPI_BIAS | EPI_RESID, false><<<g, 256, 0, stream>>>(
                o, 0, w_outT + (long)L * DIM * DIM, 0, x, 1, 0, 0,
                b_out + (long)L * DIM, x, MROWS, DIM, DIM, DIM, 1.0f);
        }
        ln_rows<true><<<MROWS, 256, 0, stream>>>(x, BIGDIV, 0, DIM,
                                                 h, BIGDIV, 0, DIM,
                                                 ff_ln_w + (long)L * DIM, ff_ln_b + (long)L * DIM, DIM);
        {   // MLP up + GELU -> bf16 hid
            dim3 g(cdiv(MLP_DIM, BN), cdiv(MROWS, BM), 1);
            gemm_bf16t<EPI_BIAS | EPI_GELU, true><<<g, 256, 0, stream>>>(
                h, 0, w1T + (long)L * DIM * MLP_DIM, 0, hid, 1, 0, 0,
                b1 + (long)L * MLP_DIM, nullptr, MROWS, MLP_DIM, DIM, MLP_DIM, 1.0f);
        }
        {   // MLP down + bias + residual -> x
            dim3 g(cdiv(DIM, BN), cdiv(MROWS, BM), 1);
            gemm_bf16t<EPI_BIAS | EPI_RESID, false><<<g, 256, 0, stream>>>(
                hid, 0, w2T + (long)L * MLP_DIM * DIM, 0, x, 1, 0, 0,
                b2 + (long)L * DIM, x, MROWS, DIM, MLP_DIM, DIM, 1.0f);
        }
    }

    // ---- head: LN(cls) @ head_w + head_b ----
    ln_rows<true><<<BATCH, 256, 0, stream>>>(x, 1, (long)SEQ * DIM, 0,
                                             h /*hcls*/, BIGDIV, 0, DIM,
                                             head_ln_w, head_ln_b, DIM);
    {
        dim3 g(cdiv(1000, BN), cdiv(BATCH, BM), 1);
        gemm_bf16t<EPI_BIAS, false><<<g, 256, 0, stream>>>(
            h, 0, head_wT, 0, out, 1, 0, 0,
            head_b, nullptr, BATCH, 1000, DIM, 1000, 1.0f);
    }
}
